// GCN_78022375899436
// MI455X (gfx1250) — matrix operand order repaired and checked by reading, not compiled, not run on hardware
//
#include <hip/hip_runtime.h>
#include <hip/hip_bf16.h>
#include <math.h>

// ---------------------------------------------------------------------------
// GCN 2-layer forward for MI455X (gfx1250, wave32, WMMA).
//
// N=50000 nodes, D_in=512, D_h=256, D_out=64, E=1.6M edges.
// - GEMMs: v_wmma_f32_16x16x32_bf16, f32 accumulate.  A and B are both
//   pre-swizzled into native WMMA fragment layouts (one v16bf / lane / frag),
//   each wave computes 2 M-tiles x NT N-tiles so every B fragment feeds two
//   WMMAs.  Layer-2's A swizzle is fused into the ReLU epilogue.
// - Edge aggregation: h1 (51MB) / h2 (12.8MB) are L2-resident (192MB L2);
//   scatter-add uses device-scope global_atomic_add_f32, with
//   global_prefetch_b8 lookahead on the gathered rows.
// ---------------------------------------------------------------------------

typedef __attribute__((ext_vector_type(16))) __bf16 v16bf;
typedef __attribute__((ext_vector_type(8)))  float  v8f;

#define NNODES 50000
#define DIN    512
#define DH     256
#define DOUT   64

__device__ __forceinline__ void atomAddF(float* p, float v) {
    __hip_atomic_fetch_add(p, v, __ATOMIC_RELAXED, __HIP_MEMORY_SCOPE_AGENT);
}

// Map K-offset within a 32-wide chunk -> (element index, lane-half) of the
// 16-bit A fragment (CDNA5 ISA "16-bit A-Matrix 16x32"):
//   elem i=2j+t holds K = (j<4) ? 8*half+2j+t : 16+8*half+2(j-4)+t
__device__ __forceinline__ int a_elem_of_k(int kk, int& half) {
    int jbase = (kk & 16) ? 4 : 0;
    int kr    = kk & 15;
    half      = kr >> 3;
    int j     = jbase + ((kr & 7) >> 1);
    return 2 * j + (kk & 1);
}

__global__ void k_zero(float* __restrict__ p, int n) {
    int i = blockIdx.x * 256 + threadIdx.x;
    if (i < n) p[i] = 0.0f;
}

__global__ void k_degree(const int* __restrict__ dst, int E, float* __restrict__ deg) {
    int e = blockIdx.x * 256 + threadIdx.x;
    if (e < E) atomAddF(&deg[dst[e]], 1.0f);
}

__global__ void k_dinv(const float* __restrict__ deg, float* __restrict__ dinv, int n) {
    int i = blockIdx.x * 256 + threadIdx.x;
    if (i < n) dinv[i] = rsqrtf(deg[i] + 1.0f);  // +1 self loop
}

// B[K,Ncols] f32 -> bf16 B-fragments: out[((tn*KC+kc)*32+lane)*16+i] where
// lane&15 = column-in-tile, K = kc*32 + (i<8 ? 8*(lane>>4)+i : 16+8*(lane>>4)+i-8).
__global__ void k_swizzle_b(const float* __restrict__ B, __bf16* __restrict__ out,
                            int K, int Ncols) {
    int idx = blockIdx.x * 256 + threadIdx.x;
    if (idx >= K * Ncols) return;
    int KC   = K >> 5;
    int i    = idx & 15;
    int lane = (idx >> 4) & 31;
    int rem  = idx >> 9;
    int kc   = rem % KC;
    int tn   = rem / KC;
    int k    = kc * 32 + ((i < 8) ? (8 * (lane >> 4) + i) : (16 + 8 * (lane >> 4) + (i - 8)));
    int col  = tn * 16 + (lane & 15);
    out[idx] = (__bf16)B[(size_t)k * Ncols + col];
}

// X[nrows,K] f32 -> bf16 A-fragments: out[((mt*KC+kc)*32+lane)*16+elem].
__global__ void k_swizzle_a(const float* __restrict__ X, __bf16* __restrict__ out,
                            int K, int nrows) {
    int idx = blockIdx.x * 256 + threadIdx.x;
    if (idx >= nrows * K) return;
    int row = idx / K;
    int k   = idx % K;
    int KC  = K >> 5;
    int kc  = k >> 5;
    int half, e = a_elem_of_k(k & 31, half);
    size_t dst = (((size_t)(row >> 4) * KC + kc) * 32 + half * 16 + (row & 15)) * 16 + e;
    out[dst] = (__bf16)X[idx];
}

// ---------------------------------------------------------------------------
// C[.,Ncols] = Asw x Bsw.  One wave computes M-tiles {2p, 2p+1} x NT N-tiles
// starting at tile ntOff = blockIdx.y*NT.  Per k-chunk: 2 A-frag loads +
// NT B-frag loads feed 2*NT WMMAs (B reused across both M-tiles).
// C/D layout: VGPR r -> M = r + 8*(lane>>4), N = lane&15.
// ---------------------------------------------------------------------------
template <int NT, int KC>
__global__ __launch_bounds__(256)
void k_gemm(const __bf16* __restrict__ Asw, const __bf16* __restrict__ Bsw,
            float* __restrict__ C, int Ncols, int mtiles, int mpairs) {
    int wave = threadIdx.x >> 5;                    // wave-uniform
    int pair = blockIdx.x * 8 + wave;
    if (pair >= mpairs) return;                     // uniform -> EXEC all-1s
    int lane  = threadIdx.x & 31;
    int half  = lane >> 4;
    int l16   = lane & 15;
    int ntOff = blockIdx.y * NT;
    int m0t   = pair * 2;
    int m1t   = m0t + 1;
    bool has2 = (m1t < mtiles);                     // wave-uniform tail guard

    const v16bf* a0 = (const v16bf*)Asw + (size_t)m0t * KC * 32 + lane;
    const v16bf* a1 = has2 ? ((const v16bf*)Asw + (size_t)m1t * KC * 32 + lane) : a0;
    const v16bf* bb = (const v16bf*)Bsw + (size_t)ntOff * KC * 32 + lane;

    v8f acc0[NT], acc1[NT];
#pragma unroll
    for (int t = 0; t < NT; ++t) { acc0[t] = (v8f){}; acc1[t] = (v8f){}; }

    for (int kc = 0; kc < KC; ++kc) {
        v16bf af0 = a0[(size_t)kc * 32];
        v16bf af1 = a1[(size_t)kc * 32];
#pragma unroll
        for (int t = 0; t < NT; ++t) {
            v16bf bf = bb[(size_t)(t * KC + kc) * 32];
            acc0[t] = __builtin_amdgcn_wmma_f32_16x16x32_bf16(
                false, af0, false, bf, (short)0, acc0[t], false, false);
            acc1[t] = __builtin_amdgcn_wmma_f32_16x16x32_bf16(
                false, af1, false, bf, (short)0, acc1[t], false, false);
        }
    }

#pragma unroll
    for (int r = 0; r < 8; ++r) {
        float* crow = C + (size_t)(m0t * 16 + r + 8 * half) * Ncols + ntOff * 16 + l16;
#pragma unroll
        for (int t = 0; t < NT; ++t) crow[t * 16] = acc0[t][r];
    }
    if (has2) {
#pragma unroll
        for (int r = 0; r < 8; ++r) {
            float* crow = C + (size_t)(m1t * 16 + r + 8 * half) * Ncols + ntOff * 16 + l16;
#pragma unroll
            for (int t = 0; t < NT; ++t) crow[t * 16] = acc1[t][r];
        }
    }
}

// One wave per edge: gather 256 f32 of h[src], scale, atomic-scatter to agg[dst].
// Prefetch the gathered row of an edge 64 ahead (global_prefetch_b8).
__global__ __launch_bounds__(256)
void k_agg256(const int* __restrict__ src, const int* __restrict__ dst,
              const float* __restrict__ dinv, const float* __restrict__ h,
              float* __restrict__ agg, int E) {
    int e = blockIdx.x * 8 + (threadIdx.x >> 5);
    if (e >= E) return;
    int lane = threadIdx.x & 31;
    int e2 = e + 64;
    if (e2 < E) {
        int s2 = src[e2];
        __builtin_prefetch(h + (size_t)s2 * DH + lane * 8, 0, 3);
    }
    int s = src[e], d = dst[e];
    float nrm = dinv[s] * dinv[d];
    const float4* hs = (const float4*)(h + (size_t)s * DH);
    float* ad = agg + (size_t)d * DH;
    float4 a = hs[lane];        // cols [0,128)
    float4 b = hs[lane + 32];   // cols [128,256)
    int o = lane * 4;
    atomAddF(ad + o + 0, a.x * nrm);
    atomAddF(ad + o + 1, a.y * nrm);
    atomAddF(ad + o + 2, a.z * nrm);
    atomAddF(ad + o + 3, a.w * nrm);
    atomAddF(ad + o + 128, b.x * nrm);
    atomAddF(ad + o + 129, b.y * nrm);
    atomAddF(ad + o + 130, b.z * nrm);
    atomAddF(ad + o + 131, b.w * nrm);
}

__global__ __launch_bounds__(256)
void k_agg64(const int* __restrict__ src, const int* __restrict__ dst,
             const float* __restrict__ dinv, const float* __restrict__ h,
             float* __restrict__ agg, int E) {
    int e = blockIdx.x * 8 + (threadIdx.x >> 5);
    if (e >= E) return;
    int lane = threadIdx.x & 31;
    int e2 = e + 64;
    if (e2 < E) {
        int s2 = src[e2];
        __builtin_prefetch(h + (size_t)s2 * DOUT + lane * 2, 0, 3);
    }
    int s = src[e], d = dst[e];
    float nrm = dinv[s] * dinv[d];
    float2 a = ((const float2*)(h + (size_t)s * DOUT))[lane];
    float* ad = agg + (size_t)d * DOUT + lane * 2;
    atomAddF(ad + 0, a.x * nrm);
    atomAddF(ad + 1, a.y * nrm);
}

// relu(agg1 + h1*dinv^2 + b1) written directly as bf16 A-fragments for GEMM2.
__global__ void k_post1(const float* __restrict__ agg, const float* __restrict__ h,
                        const float* __restrict__ dinv, const float* __restrict__ b1,
                        __bf16* __restrict__ a2sw) {
    int idx = blockIdx.x * 256 + threadIdx.x;     // NNODES*DH threads
    int i = idx >> 8;                             // DH == 256
    int c = idx & 255;
    float di = dinv[i];
    float v = agg[idx] + h[idx] * di * di + b1[c];
    v = v > 0.0f ? v : 0.0f;
    int kc = c >> 5;                              // KC = 8 for K=DH
    int half, e = a_elem_of_k(c & 31, half);
    size_t dst = (((size_t)(i >> 4) * 8 + kc) * 32 + half * 16 + (i & 15)) * 16 + e;
    a2sw[dst] = (__bf16)v;
}

// out <- log_softmax(agg2 + h2*dinv^2 + b2), one wave per node (64 cols).
__global__ __launch_bounds__(256)
void k_post2(const float* __restrict__ agg, const float* __restrict__ h,
             const float* __restrict__ dinv, const float* __restrict__ b2,
             float* __restrict__ out, int n) {
    int i = blockIdx.x * 8 + (threadIdx.x >> 5);
    if (i >= n) return;
    int lane = threadIdx.x & 31;
    float di = dinv[i];
    float sl = di * di;
    size_t base = (size_t)i * DOUT;
    int c0 = lane * 2;
    float v0 = agg[base + c0]     + h[base + c0]     * sl + b2[c0];
    float v1 = agg[base + c0 + 1] + h[base + c0 + 1] * sl + b2[c0 + 1];
    float m = fmaxf(v0, v1);
#pragma unroll
    for (int off = 16; off > 0; off >>= 1)
        m = fmaxf(m, __shfl_xor(m, off, 32));
    float s = __expf(v0 - m) + __expf(v1 - m);
#pragma unroll
    for (int off = 16; off > 0; off >>= 1)
        s += __shfl_xor(s, off, 32);
    float lse = m + __logf(s);
    out[base + c0]     = v0 - lse;
    out[base + c0 + 1] = v1 - lse;
}

extern "C" void kernel_launch(void* const* d_in, const int* in_sizes, int n_in,
                              void* d_out, int out_size, void* d_ws, size_t ws_size,
                              hipStream_t stream) {
    const float* x  = (const float*)d_in[0];   // [50000,512]
    const float* W1 = (const float*)d_in[1];   // [512,256]
    const float* b1 = (const float*)d_in[2];   // [256]
    const float* W2 = (const float*)d_in[3];   // [256,64]
    const float* b2 = (const float*)d_in[4];   // [64]
    const int*   ei = (const int*)d_in[5];     // [2,E]
    const int E = in_sizes[5] / 2;
    const int* srcp = ei;
    const int* dstp = ei + E;

    // Workspace layout (float units), ~154 MB.
    // xsw (51.2MB) is dead after GEMM1, so a2sw/h2/agg2 alias its region.
    float* ws   = (float*)d_ws;
    float* deg  = ws;                              //      50,048
    float* dinv = deg  + 50048;                    //      50,048
    float* h1   = dinv + 50048;                    //  12,800,000 (f32)
    float* agg1 = h1   + (size_t)NNODES * DH;      //  12,800,000 (f32)
    float* xreg = agg1 + (size_t)NNODES * DH;      //  12,800,000 (xsw region)
    __bf16* xsw  = (__bf16*)xreg;                  //  [0 .. 25.6M bf16)
    __bf16* a2sw = (__bf16*)xreg;                  //  aliases xsw[0 .. 12.8M bf16)
    float* h2   = xreg + 6400000;                  //   3,200,000 (f32)
    float* agg2 = h2   + (size_t)NNODES * DOUT;    //   3,200,000 (f32)
    float* wend = agg2 + (size_t)NNODES * DOUT;
    __bf16* w1sw = (__bf16*)wend;                  // 512*256 bf16
    __bf16* w2sw = (__bf16*)(wend + (DIN * DH) / 2);

    const int nAgg1 = NNODES * DH;
    const int nAgg2 = NNODES * DOUT;
    const int mtiles = NNODES / 16;                // 3125
    const int mpairs = (mtiles + 1) / 2;           // 1563
    const int gemmBlk = (mpairs + 7) / 8;          // 196

    // Degrees + normalization (zero deg first; ws is never re-zeroed).
    k_zero<<<(NNODES + 255) / 256, 256, 0, stream>>>(deg, NNODES);
    k_zero<<<(nAgg1 + 255) / 256, 256, 0, stream>>>(agg1, nAgg1);
    k_degree<<<(E + 255) / 256, 256, 0, stream>>>(dstp, E, deg);
    k_dinv<<<(NNODES + 255) / 256, 256, 0, stream>>>(deg, dinv, NNODES);

    // Operand swizzles (bf16 WMMA fragment layouts).
    k_swizzle_b<<<(DIN * DH + 255) / 256, 256, 0, stream>>>(W1, w1sw, DIN, DH);
    k_swizzle_b<<<(DH * DOUT + 255) / 256, 256, 0, stream>>>(W2, w2sw, DH, DOUT);
    k_swizzle_a<<<(NNODES * DIN + 255) / 256, 256, 0, stream>>>(x, xsw, DIN, NNODES);

    // Layer 1: h1 = x @ W1.  Waves: 2 M-tiles x 8 N-tiles; blockIdx.y = N half.
    k_gemm<8, 16><<<dim3(gemmBlk, 2), 256, 0, stream>>>(xsw, w1sw, h1, DH, mtiles, mpairs);

    // xsw now dead: safe to zero agg2 (aliased region) and overwrite with a2sw.
    k_zero<<<(nAgg2 + 255) / 256, 256, 0, stream>>>(agg2, nAgg2);

    k_agg256<<<(E + 7) / 8, 256, 0, stream>>>(srcp, dstp, dinv, h1, agg1, E);
    k_post1<<<nAgg1 / 256, 256, 0, stream>>>(agg1, h1, dinv, b1, a2sw);

    // Layer 2: h2 = relu_h1 @ W2.  Waves: 2 M-tiles x 4 N-tiles.
    k_gemm<4, 8><<<dim3(gemmBlk, 1), 256, 0, stream>>>(a2sw, w2sw, h2, DOUT, mtiles, mpairs);

    k_agg64<<<(E + 7) / 8, 256, 0, stream>>>(srcp, dstp, dinv, h2, agg2, E);
    k_post2<<<(NNODES + 7) / 8, 256, 0, stream>>>(agg2, h2, dinv, b2, (float*)d_out, NNODES);
}
